// CrowdDynamicNet_90194313216498
// MI455X (gfx1250) — matrix operand order repaired and checked by reading, not compiled, not run on hardware
//
#include <hip/hip_runtime.h>
#include <hip/hip_bf16.h>

typedef __attribute__((ext_vector_type(16))) __bf16 v16bf;
typedef __attribute__((ext_vector_type(8)))  float  v8f;
typedef __attribute__((ext_vector_type(4)))  int    v4i;

#define T_SEQ  131072
#define BATCH  6
#define IN0    32
#define HID    3
#define NL     8
#define G3     9            // 3*HID gates
#define BH     18           // BATCH*HID
#define MROWS  (T_SEQ*BATCH)   // 786432 GEMM rows
#define NTILES (MROWS/16)      // 49152
#define TPB    8               // tiles per block in GEMM
#define CHUNK  64              // timesteps per pipeline chunk
#define NCHUNK (T_SEQ/CHUNK)

// ---------------- workspace layout (bytes) ----------------
#define XG0_FLOATS   ((size_t)MROWS * G3)            // layer-0 gate preacts (T*B, 9)
#define LOUT_FLOATS  ((size_t)(NL-1) * T_SEQ * BH)   // outputs of layers 0..6
#define XG0_BYTES    (XG0_FLOATS * sizeof(float))
#define LOUT_BYTES   (LOUT_FLOATS * sizeof(float))

// gfx1250 async global->LDS path (ASYNCcnt-tracked), guarded so either
// toolchain still compiles; histogram will show whether it lowered.
#if __has_builtin(__builtin_amdgcn_global_load_async_to_lds_b128) && \
    __has_builtin(__builtin_amdgcn_s_wait_asynccnt)
#define HAVE_ASYNC_LDS 1
typedef __attribute__((address_space(1))) v4i* gptr_v4i;
typedef __attribute__((address_space(3))) v4i* lptr_v4i;
#else
#define HAVE_ASYNC_LDS 0
#endif

__device__ __forceinline__ float fast_sigmoid(float x) {
    return 1.0f / (1.0f + __expf(-x));
}
__device__ __forceinline__ float fast_tanh(float x) {
    float e = __expf(-2.0f * x);
    return (1.0f - e) / (1.0f + e);
}

// -------- kernel 0: zero the producer progress flags --------
__global__ void gru_init_flags(int* flags) {
    if (threadIdx.x < NL) flags[threadIdx.x] = 0;
}

// -------- kernel 1: layer-0 input projection via WMMA --------
// xg0[t*B+b][g] = sum_k x[t,b,k] * w_ih0[g,k] + b_ih[0][g]
// One v_wmma_f32_16x16x32_bf16 per 16-row tile (K=32 exact).
__global__ __launch_bounds__(32)
void xg0_wmma_kernel(const float* __restrict__ x,
                     const float* __restrict__ w_ih0,
                     const float* __restrict__ b_ih,
                     float* __restrict__ xg0)
{
    const int lane = threadIdx.x;
    const int n    = lane & 15;   // output column (gate) for B/C/D frags
    const int half = lane >> 4;   // K-half selector

    // B fragment: 32x16 bf16, lane holds column n; elem i = K(half*16+i).
    v16bf bfrag;
#pragma unroll
    for (int i = 0; i < 16; ++i) {
        float v = (n < G3) ? w_ih0[n * IN0 + half * 16 + i] : 0.0f;
        bfrag[i] = (__bf16)v;
    }
    // C init = bias broadcast along rows (bias depends only on gate column).
    float bias = (n < G3) ? b_ih[n] : 0.0f;
    v8f cinit;
#pragma unroll
    for (int r = 0; r < 8; ++r) cinit[r] = bias;

    const int tile0 = blockIdx.x * TPB;
    for (int ti = 0; ti < TPB; ++ti) {
        const int tile = tile0 + ti;
        const int arow = tile * 16 + n;       // A row for this lane
        const float* xr = x + (size_t)arow * IN0;
        // A fragment 16x32 bf16: lane row=n, elem i = K(half*16+i)
        float4 a0 = *(const float4*)(xr + half * 16);
        float4 a1 = *(const float4*)(xr + half * 16 + 4);
        float4 a2 = *(const float4*)(xr + half * 16 + 8);
        float4 a3 = *(const float4*)(xr + half * 16 + 12);
        v16bf afrag;
        afrag[0]=(__bf16)a0.x; afrag[1]=(__bf16)a0.y; afrag[2]=(__bf16)a0.z; afrag[3]=(__bf16)a0.w;
        afrag[4]=(__bf16)a1.x; afrag[5]=(__bf16)a1.y; afrag[6]=(__bf16)a1.z; afrag[7]=(__bf16)a1.w;
        afrag[8]=(__bf16)a2.x; afrag[9]=(__bf16)a2.y; afrag[10]=(__bf16)a2.z; afrag[11]=(__bf16)a2.w;
        afrag[12]=(__bf16)a3.x; afrag[13]=(__bf16)a3.y; afrag[14]=(__bf16)a3.z; afrag[15]=(__bf16)a3.w;

        v8f d = __builtin_amdgcn_wmma_f32_16x16x32_bf16(
            false, afrag, false, bfrag, (short)0, cinit, false, false);

        // D layout: VGPR r, lane -> row = r + 8*half, col = n.
        // Single base address; 36-byte row stride folds into store offsets.
        if (n < G3) {
            float* p = xg0 + (size_t)(tile * 16 + 8 * half) * G3 + n;
#pragma unroll
            for (int r = 0; r < 8; ++r)
                p[r * G3] = d[r];
        }
    }
}

// -------- kernel 2: 8-deep layer-pipelined GRU recurrence --------
// One block (one wave32) per layer. Producer/consumer handoff through
// L2-resident per-layer output buffers + chunk-granular release/acquire flags.
__global__ __launch_bounds__(32)
void gru_pipeline_kernel(const float* __restrict__ hxs,
                         const float* __restrict__ w_ihR,
                         const float* __restrict__ w_hh,
                         const float* __restrict__ b_ih,
                         const float* __restrict__ b_hh,
                         const float* __restrict__ xg0,
                         float* __restrict__ layer_out,   // 7 buffers of (T,B,H)
                         float* __restrict__ out,          // (T,B,H) ++ (L,B,H)
                         int* __restrict__ flags)
{
    __shared__ float lds[CHUNK * 54];   // worst case: layer-0 gate chunk

    const int layer  = blockIdx.x;
    const int lane   = threadIdx.x;
    const bool active = lane < BH;
    const int bl = active ? (lane / HID) : 0;   // batch index
    const int j  = active ? (lane % HID) : 0;   // hidden index

    // Per-lane gate-row coefficients (rows j, 3+j, 6+j of this layer).
    float whh[3][3], wih[3][3], bihv[3], bhhv[3];
#pragma unroll
    for (int g = 0; g < 3; ++g) {
        const int row = layer * G3 + g * HID + j;
        bihv[g] = b_ih[row];
        bhhv[g] = b_hh[row];
#pragma unroll
        for (int k = 0; k < HID; ++k) {
            whh[g][k] = w_hh[(size_t)row * HID + k];
            wih[g][k] = (layer > 0)
                ? w_ihR[((size_t)(layer - 1) * G3 + g * HID + j) * HID + k]
                : 0.0f;
        }
    }

    float h = active ? hxs[layer * BH + lane] : 0.0f;

    const float* srcbuf = (layer == 0)
        ? xg0
        : layer_out + (size_t)(layer - 1) * T_SEQ * BH;
    float* dstbuf = (layer == NL - 1)
        ? out
        : layer_out + (size_t)layer * T_SEQ * BH;
    const int src_stride = (layer == 0) ? 54 : BH;       // floats per timestep
    const int copy_f4    = CHUNK * src_stride / (4 * 32); // float4 per lane

    for (int c = 0; c < NCHUNK; ++c) {
        // Wait for the producer layer to publish this chunk.
        if (layer > 0) {
            const int need = (c + 1) * CHUNK;
            while (__hip_atomic_load(&flags[layer - 1], __ATOMIC_ACQUIRE,
                                     __HIP_MEMORY_SCOPE_AGENT) < need) {
                __builtin_amdgcn_s_sleep(1);
            }
        }
        // Stage chunk inputs into LDS (off the serial h-chain).
        {
            const float* src = srcbuf + (size_t)c * CHUNK * src_stride;
#if HAVE_ASYNC_LDS
            // gfx1250 LDS-DMA: no VGPR round trip, tracked by ASYNCcnt.
            for (int i = 0; i < copy_f4; ++i) {
                const int e = (lane + i * 32) * 4;   // float index of this b128
                __builtin_amdgcn_global_load_async_to_lds_b128(
                    (gptr_v4i)(src + e),
                    (lptr_v4i)(lds + e),
                    0, 0);
            }
            __builtin_amdgcn_s_wait_asynccnt(0);
#else
            const float4* s4 = (const float4*)src;
            float4* dst = (float4*)lds;
            for (int i = 0; i < copy_f4; ++i)
                dst[lane + i * 32] = s4[lane + i * 32];
#endif
        }
        __syncthreads();
        if (c + 1 < NCHUNK)
            __builtin_prefetch(srcbuf + (size_t)(c + 1) * CHUNK * src_stride, 0, 1);

        float* dptr = dstbuf + (size_t)c * CHUNK * BH + lane;
        for (int s = 0; s < CHUNK; ++s) {
            float xr, xz, xn;
            if (layer == 0) {
                const float* g = lds + s * 54 + bl * G3 + j;
                xr = g[0]; xz = g[3]; xn = g[6];
            } else {
                const float* xi = lds + s * BH + bl * HID;
                const float x0 = xi[0], x1 = xi[1], x2 = xi[2];
                xr = bihv[0] + wih[0][0] * x0 + wih[0][1] * x1 + wih[0][2] * x2;
                xz = bihv[1] + wih[1][0] * x0 + wih[1][1] * x1 + wih[1][2] * x2;
                xn = bihv[2] + wih[2][0] * x0 + wih[2][1] * x1 + wih[2][2] * x2;
            }
            // Gather h[b][0..2] from sibling lanes (uniform, all lanes exec).
            const float h0 = __shfl(h, bl * HID + 0);
            const float h1 = __shfl(h, bl * HID + 1);
            const float h2 = __shfl(h, bl * HID + 2);
            const float hr = bhhv[0] + whh[0][0] * h0 + whh[0][1] * h1 + whh[0][2] * h2;
            const float hz = bhhv[1] + whh[1][0] * h0 + whh[1][1] * h1 + whh[1][2] * h2;
            const float hn = bhhv[2] + whh[2][0] * h0 + whh[2][1] * h1 + whh[2][2] * h2;
            const float r  = fast_sigmoid(xr + hr);
            const float z  = fast_sigmoid(xz + hz);
            const float nn = fast_tanh(xn + r * hn);
            h = (1.0f - z) * nn + z * h;
            if (active)
                dptr[s * BH] = h;
        }
        __syncthreads();
        // Publish chunk to the consumer layer.
        if (layer < NL - 1 && lane == 0) {
            __hip_atomic_store(&flags[layer], (c + 1) * CHUNK,
                               __ATOMIC_RELEASE, __HIP_MEMORY_SCOPE_AGENT);
        }
    }

    // Final hidden states: d_out tail, stacked (L, B, H).
    if (active)
        out[(size_t)T_SEQ * BH + layer * BH + lane] = h;
}

extern "C" void kernel_launch(void* const* d_in, const int* in_sizes, int n_in,
                              void* d_out, int out_size, void* d_ws, size_t ws_size,
                              hipStream_t stream) {
    (void)in_sizes; (void)n_in; (void)out_size; (void)ws_size;
    const float* x     = (const float*)d_in[0];  // (T,B,I)
    const float* hxs   = (const float*)d_in[1];  // (L,B,H)
    const float* w_ih0 = (const float*)d_in[2];  // (3H,I)
    const float* w_ihR = (const float*)d_in[3];  // (L-1,3H,H)
    const float* w_hh  = (const float*)d_in[4];  // (L,3H,H)
    const float* b_ih  = (const float*)d_in[5];  // (L,3H)
    const float* b_hh  = (const float*)d_in[6];  // (L,3H)
    float* out = (float*)d_out;

    char*  ws        = (char*)d_ws;
    float* xg0       = (float*)ws;
    float* layer_out = (float*)(ws + XG0_BYTES);
    int*   flags     = (int*)(ws + XG0_BYTES + LOUT_BYTES);

    gru_init_flags<<<1, 32, 0, stream>>>(flags);
    xg0_wmma_kernel<<<NTILES / TPB, 32, 0, stream>>>(x, w_ih0, b_ih, xg0);
    gru_pipeline_kernel<<<NL, 32, 0, stream>>>(hxs, w_ihR, w_hh, b_ih, b_hh,
                                               xg0, layer_out, out, flags);
}
